// Model_flow_56745107914792
// MI455X (gfx1250) — matrix operand order repaired and checked by reading, not compiled, not run on hardware
//
#include <hip/hip_runtime.h>
#include <hip/hip_bf16.h>

#define B_ 8
#define H_ 256
#define W_ 832

typedef __attribute__((ext_vector_type(16))) _Float16 v16h;
typedef __attribute__((ext_vector_type(8)))  _Float16 v8h;
typedef __attribute__((ext_vector_type(8)))  float    v8f;
typedef __attribute__((ext_vector_type(4)))  float    v4f;

typedef __attribute__((address_space(1))) int gint_t;   // global AS
typedef __attribute__((address_space(3))) int lint_t;   // LDS AS

// ---- CDNA5 async global->LDS copy (ASYNCcnt-tracked) -----------------------
__device__ __forceinline__ void async_g2l_b32(const void* g, void* l) {
#if __has_builtin(__builtin_amdgcn_global_load_async_to_lds_b32)
    __builtin_amdgcn_global_load_async_to_lds_b32((gint_t*)g, (lint_t*)l, 0, 0);
#else
    unsigned lofs = (unsigned)(unsigned long long)(lint_t*)l;
    asm volatile("global_load_async_to_lds_b32 %0, %1, off"
                 :: "v"(lofs), "v"(g) : "memory");
#endif
}
__device__ __forceinline__ void wait_async0() {
#if __has_builtin(__builtin_amdgcn_s_wait_asynccnt)
    __builtin_amdgcn_s_wait_asynccnt(0);
#else
    asm volatile("s_wait_asynccnt 0" ::: "memory");
#endif
}

__device__ __forceinline__ void atomic_addf(float* p, float v) {
    unsafeAtomicAdd(p, v);   // native global_atomic_add_f32 on gfx1250
}

// ---- block reduction (wave32) ----------------------------------------------
__device__ __forceinline__ float block_sum256(float v, float* sm) {
    for (int o = 16; o > 0; o >>= 1) v += __shfl_down(v, o, 32);
    int lane = threadIdx.x & 31, w = threadIdx.x >> 5;
    if (lane == 0) sm[w] = v;
    __syncthreads();
    v = 0.f;
    if (w == 0) {
        v = (lane < 8) ? sm[lane] : 0.f;
        for (int o = 4; o > 0; o >>= 1) v += __shfl_down(v, o, 32);
    }
    __syncthreads();
    return v; // valid on thread 0
}

// ---- kernels ---------------------------------------------------------------
__global__ void k_zero(float* p, int n) {
    int i = blockIdx.x * blockDim.x + threadIdx.x;
    if (i < n) p[i] = 0.f;
}

// NCHW full-res images -> NHWC area-downsampled tgt/src at scale
__global__ void k_down(const float* __restrict__ i1, const float* __restrict__ i2,
                       float* __restrict__ tgt, float* __restrict__ src,
                       int Hs, int Ws, int f) {
    int i = blockIdx.x * blockDim.x + threadIdx.x;
    int n = B_ * Hs * Ws;
    if (i >= n) return;
    int x = i % Ws, y = (i / Ws) % Hs, b = i / (Ws * Hs);
    float inv = 1.f / (float)(f * f);
    for (int c = 0; c < 3; ++c) {
        const float* p1 = i1 + ((size_t)(b * 3 + c) * H_ + (size_t)y * f) * W_ + (size_t)x * f;
        const float* p2 = i2 + ((size_t)(b * 3 + c) * H_ + (size_t)y * f) * W_ + (size_t)x * f;
        float s1 = 0.f, s2 = 0.f;
        for (int fy = 0; fy < f; ++fy)
            for (int fx = 0; fx < f; ++fx) {
                s1 += p1[(size_t)fy * W_ + fx];
                s2 += p2[(size_t)fy * W_ + fx];
            }
        tgt[(size_t)i * 3 + c] = s1 * inv;
        src[(size_t)i * 3 + c] = s2 * inv;
    }
}

// forward bilinear splat of ones(flow_rev) -> occu (unclipped sums)
__global__ void k_splat(const float* __restrict__ flr, float* __restrict__ occu,
                        int Hs, int Ws) {
    int i = blockIdx.x * blockDim.x + threadIdx.x;
    int n = B_ * Hs * Ws;
    if (i >= n) return;
    int x = i % Ws, y = (i / Ws) % Hs, b = i / (Ws * Hs);
    float u = flr[(size_t)i * 2], v = flr[(size_t)i * 2 + 1];
    float xx = (float)x + u, yy = (float)y + v;
    float x0f = floorf(xx), y0f = floorf(yy);
    float fx = xx - x0f, fy = yy - y0f;
    int x0 = (int)x0f, y0 = (int)y0f;
    int xs[4] = {x0, x0, x0 + 1, x0 + 1};
    int ys[4] = {y0, y0 + 1, y0, y0 + 1};
    float wts[4] = {(1.f - fx) * (1.f - fy), (1.f - fx) * fy,
                    fx * (1.f - fy), fx * fy};
    size_t base = (size_t)b * Hs * Ws;
    for (int k = 0; k < 4; ++k) {
        int xi = xs[k], yi = ys[k];
        if (xi >= 0 && xi < Ws && yi >= 0 && yi < Hs)
            atomic_addf(&occu[base + (size_t)yi * Ws + xi], wts[k]);
    }
}

// backward warp + |proj-tgt|*occu + occu sums; emit f16 pm/tm planes (NCHW)
__global__ void k_warp(const float* __restrict__ flow, const float* __restrict__ tgt,
                       const float* __restrict__ src, const float* __restrict__ occu,
                       _Float16* __restrict__ pm, _Float16* __restrict__ tm,
                       float* accOcc, float* accErr, int Hs, int Ws) {
    __shared__ float sm1[8], sm2[8];
    int i = blockIdx.x * blockDim.x + threadIdx.x;
    int n = B_ * Hs * Ws;
    float po = 0.f, pe = 0.f;
    if (i < n) {
        int x = i % Ws, y = (i / Ws) % Hs, b = i / (Ws * Hs);
        float u = flow[(size_t)i * 2], v = flow[(size_t)i * 2 + 1];
        float fx = fminf(fmaxf((float)x + u, 0.f), (float)(Ws - 1));
        float fy = fminf(fmaxf((float)y + v, 0.f), (float)(Hs - 1));
        float x0f = floorf(fx), y0f = floorf(fy);
        float wx = fx - x0f, wy = fy - y0f;
        int x0 = (int)x0f, y0 = (int)y0f;
        int x1 = min(x0 + 1, Ws - 1), y1 = min(y0 + 1, Hs - 1);
        size_t base = (size_t)b * Hs * Ws;
        size_t ia = (base + (size_t)y0 * Ws + x0) * 3;
        size_t ib = (base + (size_t)y1 * Ws + x0) * 3;
        size_t ic = (base + (size_t)y0 * Ws + x1) * 3;
        size_t id = (base + (size_t)y1 * Ws + x1) * 3;
        float wa = (1.f - wx) * (1.f - wy), wb = (1.f - wx) * wy;
        float wc = wx * (1.f - wy),        wd = wx * wy;
        float o = fminf(fmaxf(occu[i], 0.f), 1.f);
        po = o;
        size_t plane = (size_t)Hs * Ws;
        size_t pix = (size_t)y * Ws + x;
        for (int c = 0; c < 3; ++c) {
            float p = wa * src[ia + c] + wb * src[ib + c] + wc * src[ic + c] + wd * src[id + c];
            float t = tgt[(size_t)i * 3 + c];
            pe += fabsf(p - t) * o;
            pm[(size_t)(b * 3 + c) * plane + pix] = (_Float16)(p * o);
            tm[(size_t)(b * 3 + c) * plane + pix] = (_Float16)(t * o);
        }
    }
    float s0 = block_sum256(po, sm1);
    float s1 = block_sum256(pe, sm2);
    if (threadIdx.x == 0) { atomic_addf(accOcc, s0); atomic_addf(accErr, s1); }
}

// SSIM: 1 wave per 16x16 tile (14x14 valid outputs). Async-load f16 tiles to
// LDS, horizontal 3-tap box sums via v_wmma_f32_16x16x32_f16 banded matmul,
// vertical 3-tap + SSIM formula in f32.
__global__ void k_ssim(const _Float16* __restrict__ pm, const _Float16* __restrict__ tm,
                       float* accSsim, int Hs, int Ws) {
    __shared__ alignas(16) _Float16 sX[256];
    __shared__ alignas(16) _Float16 sY[256];
    __shared__ alignas(16) float cm[5][256]; // [col][row] horizontal sums
    int lane = threadIdx.x;
    int ty0 = blockIdx.y * 14, tx0 = blockIdx.x * 14;
    const _Float16* gx = pm + (size_t)blockIdx.z * Hs * Ws;
    const _Float16* gy = tm + (size_t)blockIdx.z * Hs * Ws;

    // stage 16x16 f16 tiles: 4 async b32 per map (32 lanes x 4B = 4 rows/op)
    for (int it = 0; it < 4; ++it) {
        int r = it * 4 + (lane >> 3);
        int cp = (lane & 7) * 2;
        int grow = min(ty0 + r, Hs - 1);
        int gcol = min(tx0 + cp, Ws - 2);
        size_t go = (size_t)grow * Ws + gcol;
        async_g2l_b32(gx + go, &sX[r * 16 + cp]);
        async_g2l_b32(gy + go, &sY[r * 16 + cp]);
    }
    wait_async0();

    // A fragments (16-bit A 16x32 layout): lane holds row (lane&15),
    // K = (lane>>4)*8 .. +7 in elems 0..7; K>=16 half is zero-padding.
    int row = lane & 15;
    int kb  = (lane >> 4) * 8;
    v8h tx = *(const v8h*)&sX[row * 16 + kb];
    v8h ty = *(const v8h*)&sY[row * 16 + kb];
    v16h ax, ay;
    for (int j = 0; j < 16; ++j) { ax[j] = (_Float16)0.f; ay[j] = (_Float16)0.f; }
    for (int j = 0; j < 8; ++j)  { ax[j] = tx[j]; ay[j] = ty[j]; }
    v16h axx = ax * ax, ayy = ay * ay, axy = ax * ay;

    // B fragment: tri-diagonal band S[k][n] = (|k-n|<=1), rows 16..31 zero.
    int nn = lane & 15;
    v16h bS;
    for (int k = 0; k < 16; ++k) {
        int d = kb + k - nn;
        bS[k] = (_Float16)(((d * d <= 1) && (k < 8)) ? 1.f : 0.f);
    }

    v8f c0 = {};
    v8f hx  = __builtin_amdgcn_wmma_f32_16x16x32_f16(false, ax,  false, bS, (short)0, c0, false, false);
    v8f hy  = __builtin_amdgcn_wmma_f32_16x16x32_f16(false, ay,  false, bS, (short)0, c0, false, false);
    v8f hxx = __builtin_amdgcn_wmma_f32_16x16x32_f16(false, axx, false, bS, (short)0, c0, false, false);
    v8f hyy = __builtin_amdgcn_wmma_f32_16x16x32_f16(false, ayy, false, bS, (short)0, c0, false, false);
    v8f hxy = __builtin_amdgcn_wmma_f32_16x16x32_f16(false, axy, false, bS, (short)0, c0, false, false);

    // D layout: elem j = H[m0+j][coln]; store column-major to LDS (2x b128)
    int coln = lane & 15;
    int m0   = (lane >> 4) * 8;
#define STORE_CM(q, vec) do { float* dp = &cm[q][coln * 16 + m0];              \
        *(v4f*)dp       = __builtin_shufflevector(vec, vec, 0, 1, 2, 3);       \
        *(v4f*)(dp + 4) = __builtin_shufflevector(vec, vec, 4, 5, 6, 7); } while (0)
    STORE_CM(0, hx); STORE_CM(1, hy); STORE_CM(2, hxx); STORE_CM(3, hyy); STORE_CM(4, hxy);
#undef STORE_CM

    // vertical 3-tap + SSIM over 14x14 valid centers (same-wave DS ordering)
    const float C1 = 1e-4f, C2 = 9e-4f, inv9 = 1.f / 9.f;
    float vs = 0.f;
    for (int idx = lane; idx < 196; idx += 32) {
        int oy = idx / 14 + 1, ox = idx % 14 + 1;
        int gyc = ty0 + oy, gxc = tx0 + ox;
        if (gyc <= Hs - 2 && gxc <= Ws - 2) {
            int cb = ox * 16 + oy;
            float mx  = (cm[0][cb - 1] + cm[0][cb] + cm[0][cb + 1]) * inv9;
            float my  = (cm[1][cb - 1] + cm[1][cb] + cm[1][cb + 1]) * inv9;
            float exx = (cm[2][cb - 1] + cm[2][cb] + cm[2][cb + 1]) * inv9;
            float eyy = (cm[3][cb - 1] + cm[3][cb] + cm[3][cb + 1]) * inv9;
            float exy = (cm[4][cb - 1] + cm[4][cb] + cm[4][cb + 1]) * inv9;
            float sx = exx - mx * mx, sy = eyy - my * my, sxy = exy - mx * my;
            float nu = (2.f * mx * my + C1) * (2.f * sxy + C2);
            float de = (mx * mx + my * my + C1) * (sx + sy + C2);
            vs += fminf(fmaxf((1.f - nu / de) * 0.5f, 0.f), 1.f);
        }
    }
    for (int o = 16; o > 0; o >>= 1) vs += __shfl_down(vs, o, 32);
    if (lane == 0) atomic_addf(accSsim, vs);
}

// edge-aware 2nd-order smoothness sums (x and y parts separately)
__global__ void k_smooth(const float* __restrict__ flow, const float* __restrict__ tgt,
                         float* accX, float* accY, int Hs, int Ws) {
    __shared__ float sm1[8], sm2[8];
    int i = blockIdx.x * blockDim.x + threadIdx.x;
    int n = B_ * Hs * Ws;
    float px = 0.f, py = 0.f;
    if (i < n) {
        int x = i % Ws, y = (i / Ws) % Hs;
        size_t t0 = (size_t)i * 3;
        size_t f0 = (size_t)i * 2;
        if (x < Ws - 2) {
            float g = fabsf(tgt[t0 + 6] - tgt[t0 + 3]) +
                      fabsf(tgt[t0 + 7] - tgt[t0 + 4]) +
                      fabsf(tgt[t0 + 8] - tgt[t0 + 5]);
            float wx = __expf(-10.f * g * (1.f / 3.f));
            float d0 = flow[f0 + 4] - 2.f * flow[f0 + 2] + flow[f0 + 0];
            float d1 = flow[f0 + 5] - 2.f * flow[f0 + 3] + flow[f0 + 1];
            px = wx * (fabsf(d0) + fabsf(d1)) * 0.05f; // /20
        }
        if (y < Hs - 2) {
            size_t r1 = t0 + (size_t)Ws * 3, r2 = t0 + (size_t)Ws * 6;
            float g = fabsf(tgt[r2 + 0] - tgt[r1 + 0]) +
                      fabsf(tgt[r2 + 1] - tgt[r1 + 1]) +
                      fabsf(tgt[r2 + 2] - tgt[r1 + 2]);
            float wy = __expf(-10.f * g * (1.f / 3.f));
            size_t q1 = f0 + (size_t)Ws * 2, q2 = f0 + (size_t)Ws * 4;
            float d0 = flow[q2 + 0] - 2.f * flow[q1 + 0] + flow[f0 + 0];
            float d1 = flow[q2 + 1] - 2.f * flow[q1 + 1] + flow[f0 + 1];
            py = wy * (fabsf(d0) + fabsf(d1)) * 0.05f;
        }
    }
    float s0 = block_sum256(px, sm1);
    float s1 = block_sum256(py, sm2);
    if (threadIdx.x == 0) { atomic_addf(accX, s0); atomic_addf(accY, s1); }
}

__global__ void k_final(const float* __restrict__ acc, float* __restrict__ out) {
    float loss = 0.f;
    for (int s = 0; s < 4; ++s) {
        int Hs = H_ >> s, Ws = W_ >> s;
        double N1 = (double)B_ * Hs * Ws;
        float oavg  = (float)((double)acc[s] / N1) + 1e-12f;
        float errm  = (float)((double)acc[4 + s] / (N1 * 3.0));
        float ssimm = (float)((double)acc[8 + s] / ((double)B_ * 3.0 * (Hs - 2) * (Ws - 2)));
        loss += (1.0f - 0.85f) * errm / oavg + 0.85f * ssimm / oavg;
        float sx = (float)((double)acc[12 + s] / ((double)B_ * 2.0 * Hs * (Ws - 2)));
        float sy = (float)((double)acc[16 + s] / ((double)B_ * 2.0 * (Hs - 2) * Ws));
        loss += 10.0f * 0.5f * (sx + sy);
    }
    out[0] = loss;
}

// ---- host ------------------------------------------------------------------
extern "C" void kernel_launch(void* const* d_in, const int* in_sizes, int n_in,
                              void* d_out, int out_size, void* d_ws, size_t ws_size,
                              hipStream_t stream) {
    (void)in_sizes; (void)n_in; (void)out_size; (void)ws_size;
    const float* img1 = (const float*)d_in[0];
    const float* img2 = (const float*)d_in[1];

    const size_t IMG = (size_t)B_ * 3 * H_ * W_;   // 5,111,808
    const size_t PIX = (size_t)B_ * H_ * W_;       // 1,703,936
    float* ws   = (float*)d_ws;
    float* acc  = ws;
    float* tgt  = ws + 64;
    float* src  = tgt + IMG;
    float* occu = src + IMG;
    _Float16* pm = (_Float16*)(occu + PIX);
    _Float16* tm = pm + IMG;

    k_zero<<<1, 64, 0, stream>>>(acc, 64);

    for (int s = 0; s < 4; ++s) {
        int Hs = H_ >> s, Ws = W_ >> s, f = 1 << s;
        int n1 = B_ * Hs * Ws;
        int g1 = (n1 + 255) / 256;

        k_down <<<g1, 256, 0, stream>>>(img1, img2, tgt, src, Hs, Ws, f);
        k_zero <<<g1, 256, 0, stream>>>(occu, n1);
        k_splat<<<g1, 256, 0, stream>>>((const float*)d_in[6 + s], occu, Hs, Ws);
        k_warp <<<g1, 256, 0, stream>>>((const float*)d_in[2 + s], tgt, src, occu,
                                        pm, tm, acc + s, acc + 4 + s, Hs, Ws);
        dim3 gs((unsigned)((Ws - 2 + 13) / 14), (unsigned)((Hs - 2 + 13) / 14),
                (unsigned)(B_ * 3));
        k_ssim<<<gs, 32, 0, stream>>>(pm, tm, acc + 8 + s, Hs, Ws);
        k_smooth<<<g1, 256, 0, stream>>>((const float*)d_in[2 + s], tgt,
                                         acc + 12 + s, acc + 16 + s, Hs, Ws);
    }
    k_final<<<1, 1, 0, stream>>>(acc, (float*)d_out);
}
